// ScaledDotProductionAttention_47871705481475
// MI455X (gfx1250) — matrix operand order repaired and checked by reading, not compiled, not run on hardware
//
#include <hip/hip_runtime.h>

typedef __attribute__((ext_vector_type(16))) _Float16 v16h;
typedef __attribute__((ext_vector_type(8)))  _Float16 v8h;
typedef __attribute__((ext_vector_type(8)))  float    v8f;
typedef __attribute__((ext_vector_type(4)))  float    f32x4;

static constexpr int SEQ = 2048;   // S
static constexpr int DIM = 64;     // D
static constexpr int NBH = 32;     // B*H
static constexpr int TM  = 16;     // query rows per workgroup
static constexpr int NW  = 4;      // waves per block (128 threads, wave32)
static constexpr int CPW = SEQ / NW; // score columns per wave = 512

// Load 16 contiguous f32 and convert to f16 (B-fragment helper: lane holds K=j..j+15).
__device__ __forceinline__ v16h cvt16h(const float* __restrict__ p) {
  f32x4 x0 = *(const f32x4*)(p);
  f32x4 x1 = *(const f32x4*)(p + 4);
  f32x4 x2 = *(const f32x4*)(p + 8);
  f32x4 x3 = *(const f32x4*)(p + 12);
  v16h r;
#pragma unroll
  for (int j = 0; j < 4; ++j) {
    r[j]      = (_Float16)x0[j];
    r[4 + j]  = (_Float16)x1[j];
    r[8 + j]  = (_Float16)x2[j];
    r[12 + j] = (_Float16)x3[j];
  }
  return r;
}

// A-fragment for 16x16x32 f16 WMMA from a 32-float K-chunk of one Q row.
// lane<16: elems 0..15 = K0..15 ; lane>=16: K16..31.
__device__ __forceinline__ v16h load_a_f32(const float* __restrict__ p, int hi) {
  const float* p0 = p + hi * 16;
  f32x4 x0 = *(const f32x4*)(p0);
  f32x4 x1 = *(const f32x4*)(p0 + 4);
  f32x4 y0 = *(const f32x4*)(p0 + 8);
  f32x4 y1 = *(const f32x4*)(p0 + 12);
  v16h r;
#pragma unroll
  for (int j = 0; j < 4; ++j) {
    r[j]      = (_Float16)x0[j];
    r[4 + j]  = (_Float16)x1[j];
    r[8 + j]  = (_Float16)y0[j];
    r[12 + j] = (_Float16)y1[j];
  }
  return r;
}

__global__ __launch_bounds__(128) void attn_fused(
    const float* __restrict__ q, const float* __restrict__ k,
    const float* __restrict__ v, const int* __restrict__ masks,
    float* __restrict__ out, float* __restrict__ attn) {
  // 16 x 2048 masked, scaled scores in f16 = 64 KB LDS (reused by softmax + PV matmul)
  __shared__ _Float16 sdata[TM * SEQ];

  const int tid  = threadIdx.x;
  const int lane = tid & 31;
  const int wave = tid >> 5;
  const int hi   = lane >> 4;   // lane-half selector for WMMA fragments
  const int ln   = lane & 15;

  const int bh   = blockIdx.y;          // fused batch*head
  const int bb   = bh >> 4;             // batch (H = 16)
  const int row0 = blockIdx.x * TM;     // first query row of this block

  const float* qbase = q + (size_t)bh * SEQ * DIM;
  const float* kbase = k + (size_t)bh * SEQ * DIM;
  const float* vbase = v + (size_t)bh * SEQ * DIM;
  const int*   mbase = masks + (size_t)bb * SEQ * SEQ;

  // ---------------- Phase 1: scores = (Q K^T) * 0.125, masked, -> LDS f16 ----
  const float* qrow = qbase + (size_t)(row0 + ln) * DIM;
  const v16h a0 = load_a_f32(qrow,      hi);   // K = 0..31
  const v16h a1 = load_a_f32(qrow + 32, hi);   // K = 32..63

  for (int t = 0; t < 32; ++t) {
    const int col0 = wave * CPW + t * 16;
    const float* krow = kbase + (size_t)(col0 + ln) * DIM;
    if (t + 1 < 32)
      __builtin_prefetch(kbase + (size_t)(col0 + 16 + ln) * DIM, 0, 1);

    // B fragment: lane n holds column (col0+ln); halves split K 0..15 / 16..31.
    v16h b0 = cvt16h(krow + hi * 16);        // K chunk 0
    v16h b1 = cvt16h(krow + 32 + hi * 16);   // K chunk 1

    v8f c = {};
    c = __builtin_amdgcn_wmma_f32_16x16x32_f16(false, a0, false, b0, (short)0, c, false, false);
    c = __builtin_amdgcn_wmma_f32_16x16x32_f16(false, a1, false, b1, (short)0, c, false, false);

    const int colg = col0 + ln;
#pragma unroll
    for (int r = 0; r < 8; ++r) {
      const int rl = r + hi * 8;            // local row of C element r
      const int mv = mbase[(size_t)(row0 + rl) * SEQ + colg];
      const float sv = mv ? c[r] * 0.125f : -1e-12f;  // faithful tiny mask fill
      sdata[rl * SEQ + colg] = (_Float16)sv;
    }
  }
  __syncthreads();

  // ---------------- Phase 2: attention = softmax(scores) -> global ----------
  float* attn_rowbase = attn + ((size_t)bh * SEQ + row0) * SEQ;
#pragma unroll 1
  for (int rr = 0; rr < 4; ++rr) {
    const int rl = wave * 4 + rr;
    const _Float16* srow = sdata + rl * SEQ;

    float mx = -3.4e38f;
    for (int c0 = lane; c0 < SEQ; c0 += 32) mx = fmaxf(mx, (float)srow[c0]);
#pragma unroll
    for (int off = 16; off > 0; off >>= 1) mx = fmaxf(mx, __shfl_xor(mx, off, 32));

    float sum = 0.f;
    for (int c0 = lane; c0 < SEQ; c0 += 32) sum += __expf((float)srow[c0] - mx);
#pragma unroll
    for (int off = 16; off > 0; off >>= 1) sum += __shfl_xor(sum, off, 32);

    const float inv = 1.0f / sum;
    float* arow = attn_rowbase + (size_t)rl * SEQ;
    for (int c0 = lane * 4; c0 < SEQ; c0 += 128) {  // coalesced float4 stores
      f32x4 o;
#pragma unroll
      for (int j = 0; j < 4; ++j) o[j] = __expf((float)srow[c0 + j] - mx) * inv;
      *(f32x4*)(arow + c0) = o;
    }
  }

  // ---------------- Phase 3: output = masked_scores @ V (faithful bug) ------
  const int dcol0 = wave * 16;              // each wave owns 16 of the 64 d-cols
  v8f acc = {};
  const _Float16* srow = sdata + ln * SEQ;  // A: row per lane
  for (int kc = 0; kc < 64; ++kc) {
    const int k0 = kc * 32;
    // A fragment straight from LDS (16B vector loads -> ds_load_b128)
    v8h lo = *(const v8h*)(srow + k0 + hi * 16);
    v8h hh = *(const v8h*)(srow + k0 + hi * 16 + 8);
    v16h a = __builtin_shufflevector(lo, hh, 0, 1, 2, 3, 4, 5, 6, 7,
                                             8, 9, 10, 11, 12, 13, 14, 15);
    // B fragment: V rows k0+hi*16+j, column dcol0+ln
    v16h bm;
    const float* vp = vbase + (size_t)(k0 + hi * 16) * DIM + dcol0 + ln;
#pragma unroll
    for (int j = 0; j < 16; ++j) bm[j] = (_Float16)vp[(size_t)j * DIM];

    acc = __builtin_amdgcn_wmma_f32_16x16x32_f16(false, a, false, bm, (short)0, acc, false, false);
  }

  float* obase = out + ((size_t)bh * SEQ + row0) * DIM + dcol0 + ln;
#pragma unroll
  for (int r = 0; r < 8; ++r)
    obase[(size_t)(r + hi * 8) * DIM] = acc[r];
}

extern "C" void kernel_launch(void* const* d_in, const int* in_sizes, int n_in,
                              void* d_out, int out_size, void* d_ws, size_t ws_size,
                              hipStream_t stream) {
  const float* q     = (const float*)d_in[0];
  const float* k     = (const float*)d_in[1];
  const float* v     = (const float*)d_in[2];
  const int*   masks = (const int*)d_in[3];

  float* out  = (float*)d_out;                      // [2,16,2048,64]
  float* attn = out + (size_t)NBH * SEQ * DIM;      // [2,16,2048,2048] after it

  dim3 grid(SEQ / TM, NBH);   // 128 row-blocks x 32 heads
  attn_fused<<<grid, NW * 32, 0, stream>>>(q, k, v, masks, out, attn);
}
